// SWA_SinkMeta_3350074491048
// MI455X (gfx1250) — matrix operand order repaired and checked by reading, not compiled, not run on hardware
//
#include <hip/hip_runtime.h>
#include <math.h>

#define NUM_HEADS 32
#define NUM_KV_HEADS 8
#define HEAD_DIM 128
#define WINDOW 512
#define NUM_META 4

typedef __attribute__((ext_vector_type(16))) _Float16 v16h;
typedef __attribute__((ext_vector_type(8)))  _Float16 v8h;
typedef __attribute__((ext_vector_type(8)))  float    v8f;

__device__ __forceinline__ v16h concat8(v8h lo, v8h hi) {
    v16h r;
#pragma unroll
    for (int i = 0; i < 8; ++i) { r[i] = lo[i]; r[i + 8] = hi[i]; }
    return r;
}

// One 32-wide KV chunk of flash attention for one wave's 16-row Q stripe.
// MODE: 0 = interior (provably unmasked), 1 = full sink+window+causal mask,
//       2 = diagonal (causal-only mask).
template <int MODE>
__device__ __forceinline__ void process_chunk(
    int j0, int qr, int lh, int sel,
    const _Float16* __restrict__ Ksh, const _Float16* __restrict__ VTsh,
    _Float16* __restrict__ Pw,
    const v16h (&aq)[4], v8f (&o)[8], float (&m)[8], float (&l)[8]) {
    const float scale = 0.08838834764831845f;  // 1/sqrt(128)

    // ---- scores S = Q * K^T : two 16x16 tiles covering kv [j0, j0+32)
    v8f sacc0 = {}, sacc1 = {};
#pragma unroll
    for (int dc = 0; dc < 4; ++dc) {
        v16h b0 = *(const v16h*)(Ksh + lh * HEAD_DIM + dc * 32 + sel * 16);
        sacc0 = __builtin_amdgcn_wmma_f32_16x16x32_f16(
            false, aq[dc], false, b0, (short)0, sacc0, false, false);
    }
#pragma unroll
    for (int dc = 0; dc < 4; ++dc) {
        v16h b1 = *(const v16h*)(Ksh + (16 + lh) * HEAD_DIM + dc * 32 + sel * 16);
        sacc1 = __builtin_amdgcn_wmma_f32_16x16x32_f16(
            false, aq[dc], false, b1, (short)0, sacc1, false, false);
    }

    // ---- scale (+ branchless mask only where a row can actually be masked)
    float s0[8], s1[8];
#pragma unroll
    for (int r = 0; r < 8; ++r) {
        float va = sacc0[r] * scale;
        float vb = sacc1[r] * scale;
        if (MODE != 0) {
            const int i  = qr + r + sel * 8;
            const int ja = j0 + lh;
            const int jb = j0 + 16 + lh;
            int oka, okb;
            if (MODE == 2) {           // diagonal: window/sink can't trigger
                oka = (int)(i >= ja);
                okb = (int)(i >= jb);
            } else {                   // chunk 0 / window edge: full mask
                // bitwise combine (no short-circuit) -> v_cmp + v_cndmask,
                // no divergent EXEC manipulation
                oka = (int)(i >= ja) & ((int)((i - ja) <= WINDOW) | (int)(ja < NUM_META));
                okb = (int)(i >= jb) & ((int)((i - jb) <= WINDOW) | (int)(jb < NUM_META));
            }
            va = oka ? va : -1e30f;
            vb = okb ? vb : -1e30f;
        }
        s0[r] = va;
        s1[r] = vb;
    }

    // ---- online softmax: reduce across the 16 lanes holding each row
#pragma unroll
    for (int r = 0; r < 8; ++r) {
        float rmax = fmaxf(s0[r], s1[r]);
#pragma unroll
        for (int off = 1; off < 16; off <<= 1)
            rmax = fmaxf(rmax, __shfl_xor(rmax, off, 32));
        const float mn    = fmaxf(m[r], rmax);
        const float alpha = __expf(m[r] - mn);
        const float p0 = __expf(s0[r] - mn);
        const float p1 = __expf(s1[r] - mn);
        float rs = p0 + p1;
#pragma unroll
        for (int off = 1; off < 16; off <<= 1)
            rs += __shfl_xor(rs, off, 32);
        l[r] = l[r] * alpha + rs;
        m[r] = mn;
#pragma unroll
        for (int t = 0; t < 8; ++t) o[t][r] *= alpha;  // rescale O rows
        // stage P (f16) to per-wave LDS scratch in C-layout positions
        const int rowM = r + sel * 8;
        Pw[rowM * 32 + lh]      = (_Float16)p0;
        Pw[rowM * 32 + 16 + lh] = (_Float16)p1;
    }

    // ---- O += P(16x32) * V(32x128): read P back in A-matrix layout
    v16h ap = concat8(*(const v8h*)(Pw + lh * 32 + sel * 16),
                      *(const v8h*)(Pw + lh * 32 + 8 + sel * 16));
#pragma unroll
    for (int dt = 0; dt < 8; ++dt) {
        v16h bv = *(const v16h*)(VTsh + (dt * 16 + lh) * 32 + sel * 16);
        o[dt] = __builtin_amdgcn_wmma_f32_16x16x32_f16(
            false, ap, false, bv, (short)0, o[dt], false, false);
    }
}

// Flash attention with sliding-window + sink mask, GQA 4:1.
// One block = one head x 128 query rows. 8 waves, each wave owns a 16-row
// Q stripe and a full 16x128 f32 accumulator (8 WMMA C tiles).
__global__ __launch_bounds__(256)
void swa_sink_fa(const float* __restrict__ Qg, const float* __restrict__ Kg,
                 const float* __restrict__ Vg, float* __restrict__ Og, int S) {
    // K chunk (32 x 128, row major f16): B-fragments for Q*K^T are contiguous.
    // V chunk transposed (128 x 32 f16): B-fragments for P*V are contiguous.
    // P scratch: per-wave 16x32 f16 region for the C-layout -> A-layout shuffle.
    __shared__ __align__(32) _Float16 Ksh[32 * HEAD_DIM];
    __shared__ __align__(32) _Float16 VTsh[HEAD_DIM * 32];
    __shared__ __align__(32) _Float16 Psh[8 * 16 * 32];

    const int h    = blockIdx.y;
    const int kh   = h / (NUM_HEADS / NUM_KV_HEADS);
    const int q0   = blockIdx.x * 128;
    const int wave = threadIdx.x >> 5;
    const int lane = threadIdx.x & 31;
    const int lh   = lane & 15;   // column / row-in-16 within half-wave
    const int sel  = lane >> 4;   // half-wave select (ISA fragment layouts)
    const int qr   = q0 + wave * 16;

    const int QLD = NUM_HEADS * HEAD_DIM;      // 4096
    const int KLD = NUM_KV_HEADS * HEAD_DIM;   // 1024

    // ---- Q fragments in A-matrix layout (16x32 f16), 4 chunks over D=128.
    // lane<16: row lh, K = {0..15}; lane>=16: row lh, K = {16..31}
    v16h aq[4];
    {
        const float* qrow = Qg + (size_t)(qr + lh) * QLD + h * HEAD_DIM;
#pragma unroll
        for (int dc = 0; dc < 4; ++dc) {
            const int b0 = dc * 32 + sel * 16;
            const int b1 = dc * 32 + 8 + sel * 16;
            v16h a;
#pragma unroll
            for (int u = 0; u < 8; ++u) {
                a[u]     = (_Float16)qrow[b0 + u];
                a[u + 8] = (_Float16)qrow[b1 + u];
            }
            aq[dc] = a;
        }
    }

    v8f o[8] = {};                 // 16x128 f32 output accumulator
    float m[8], l[8];              // online-softmax row stats (rows r + 8*sel)
#pragma unroll
    for (int r = 0; r < 8; ++r) { m[r] = -1e30f; l[r] = 0.0f; }

    int jend = q0 + 128;           // causal upper bound for this block
    if (jend > S) jend = S;
    _Float16* Pw = Psh + wave * (16 * 32);

    // Cooperative K/V chunk load: 256 threads x 16 contiguous f32 each.
    auto loadKV = [&](int j0) {
        const int t  = threadIdx.x;
        const int jr = (t * 16) >> 7;    // kv row 0..31
        const int c0 = (t * 16) & 127;   // d offset, multiple of 16
        const float* kp = Kg + (size_t)(j0 + jr) * KLD + kh * HEAD_DIM + c0;
        const float* vp = Vg + (size_t)(j0 + jr) * KLD + kh * HEAD_DIM + c0;
        _Float16* kd = Ksh + jr * HEAD_DIM + c0;
#pragma unroll
        for (int u = 0; u < 16; u += 2) {   // packed f32->f16 (v_cvt_pk)
            auto pk = __builtin_amdgcn_cvt_pkrtz(kp[u], kp[u + 1]);
            *(decltype(pk)*)(kd + u) = pk;
        }
#pragma unroll
        for (int u = 0; u < 16; ++u)
            VTsh[(c0 + u) * 32 + jr] = (_Float16)vp[u];
    };

    // Interior chunks [bstart, q0-32] are provably unmasked for every row:
    // worst case i-j = (q0+127) - (q0-384) = 511 <= WINDOW, and j < q0 <= i.
    const int bstart = (q0 - 384 > 32) ? (q0 - 384) : 32;

    // Phase A: chunk 0 (sinks) + window-edge chunks -> full mask.
    {
        const int aend = (jend < bstart) ? jend : bstart;
        for (int j0 = 0; j0 < aend; j0 += 32) {
            if (j0 != 0 && (j0 + 32) <= (q0 - WINDOW)) continue;  // below window
            __syncthreads();
            loadKV(j0);
            __syncthreads();
            process_chunk<1>(j0, qr, lh, sel, Ksh, VTsh, Pw, aq, o, m, l);
        }
    }
    // Phase B: interior chunks -> no masking at all.
    for (int j0 = bstart; j0 + 32 <= q0; j0 += 32) {
        __syncthreads();
        loadKV(j0);
        __syncthreads();
        process_chunk<0>(j0, qr, lh, sel, Ksh, VTsh, Pw, aq, o, m, l);
    }
    // Phase C: diagonal chunks -> causal-only mask.
    for (int j0 = (q0 > 32 ? q0 : 32); j0 < jend; j0 += 32) {
        __syncthreads();
        loadKV(j0);
        __syncthreads();
        process_chunk<2>(j0, qr, lh, sel, Ksh, VTsh, Pw, aq, o, m, l);
    }

    // ---- finalize: divide by row sums, store fp32 [S, H*D]
    float inv[8];
#pragma unroll
    for (int r = 0; r < 8; ++r) inv[r] = 1.0f / l[r];
#pragma unroll
    for (int dt = 0; dt < 8; ++dt) {
#pragma unroll
        for (int r = 0; r < 8; ++r) {
            const int i = qr + r + sel * 8;
            Og[(size_t)i * QLD + h * HEAD_DIM + dt * 16 + lh] = o[dt][r] * inv[r];
        }
    }
}

extern "C" void kernel_launch(void* const* d_in, const int* in_sizes, int n_in,
                              void* d_out, int out_size, void* d_ws, size_t ws_size,
                              hipStream_t stream) {
    const float* Q = (const float*)d_in[0];
    const float* K = (const float*)d_in[1];
    const float* V = (const float*)d_in[2];
    float* O = (float*)d_out;
    const int S = in_sizes[0] / (NUM_HEADS * HEAD_DIM);   // 2048
    dim3 grid(S / 128, NUM_HEADS);
    swa_sink_fa<<<grid, dim3(256), 0, stream>>>(Q, K, V, O, S);
}
